// WindowAttentionAugmented_60550448939747
// MI455X (gfx1250) — matrix operand order repaired and checked by reading, not compiled, not run on hardware
//
#include <hip/hip_runtime.h>

#define Tt 8
#define Hh 128
#define Ww 128
#define Cc 32
#define NWIN (Tt*32*32)

typedef __attribute__((ext_vector_type(16))) _Float16 v16h;
typedef __attribute__((ext_vector_type(8)))  float    v8f;

// ---- LDS layout (bytes), regions time-multiplexed ----
// R1 [0, 33280):  ph  f16 64x40 @0      = 5120   (phases 0b-1)
//                 wq  f16 96x40 @5120   = 7680   (phases 0-1)
//                 st  f32 64x32 @12800  = 8192   (phase 0, async staging)
//                 af  f32 2x64x65 @0    = 33280  (phases 2-3)
//                 avh f16 64x40 @0      = 5120   (phases 4-5)
// R2 [33280, 41856): qk f16 64x67 = 8576        (phases 1-3; q cols 0-31, k cols 32-63)
// R2b[41856, 46208): vT f16 32x68 = 4352        (phases 1-4; V transposed [ch][tok])
// R3 [46208, 63104): ah f16 2x64x66 = 16896     (phases 3-4)
// R4 [63104, 65408): pw f16 32x36 = 2304        (phases 0-5)
#define OFF_PH   0
#define OFF_WQ   5120
#define OFF_ST   12800
#define OFF_AF   0
#define OFF_AVH  0
#define OFF_QK   33280
#define OFF_VT   41856
#define OFF_AH   46208
#define OFF_PW   63104
#define SMEM_BYTES 65408

// workspace layout: cnt f32[H*W] @0 (64KB), wq16 f16[96*32] @65536 (6144B)
#define WS_WQ16  65536

__device__ __forceinline__ int reflect_idx(int r) { return (r >= Hh) ? (2*(Hh-1) - r) : r; }

// CDNA5 async global->LDS copy, 16B per lane, GVS addressing (SGPR base + VGPR offset).
// Tracked by ASYNCcnt (CDNA5 ISA).
__device__ __forceinline__ void async_copy16(unsigned lds_byte_addr, const void* gbase, int gbyte_off) {
    asm volatile("global_load_async_to_lds_b128 %0, %1, %2"
                 :: "v"(lds_byte_addr), "v"(gbyte_off), "s"(gbase)
                 : "memory");
}
__device__ __forceinline__ void wait_async0() {
    asm volatile("s_wait_asynccnt 0x0" ::: "memory");
}

__global__ __launch_bounds__(128)
void win_attn_kernel(const float* __restrict__ x,
                     const float* __restrict__ qkv_b,
                     const float* __restrict__ table,
                     const float* __restrict__ proj_w,
                     const float* __restrict__ proj_b,
                     const int*   __restrict__ rindex,
                     const _Float16* __restrict__ wq16,   // pre-converted qkv_w (96x32 f16)
                     float* __restrict__ out,
                     float* __restrict__ cnt)
{
    extern __shared__ char smem[];
    _Float16* ph  = (_Float16*)(smem + OFF_PH);
    _Float16* wq  = (_Float16*)(smem + OFF_WQ);
    float*    stg = (float*)   (smem + OFF_ST);
    float*    af  = (float*)   (smem + OFF_AF);
    _Float16* avh = (_Float16*)(smem + OFF_AVH);
    _Float16* qk  = (_Float16*)(smem + OFF_QK);
    _Float16* vT  = (_Float16*)(smem + OFF_VT);
    _Float16* ah  = (_Float16*)(smem + OFF_AH);
    _Float16* pw  = (_Float16*)(smem + OFF_PW);

    const unsigned lds_base = (unsigned)(size_t)(void*)smem;   // flat low 32 bits == LDS byte addr

    const int tid  = threadIdx.x;
    // wave id is uniform across the wave: pin it to an SGPR so tile indices
    // (and the q/k/v epilogue split) become scalar control flow.
    const int wave = __builtin_amdgcn_readfirstlane(tid >> 5);
    const int lane = tid & 31;
    const int hv   = lane >> 4;
    const int lm   = lane & 15;

    const int blk = blockIdx.x;
    const int t  = blk >> 10;
    const int wy = (blk >> 5) & 31;
    const int wx = blk & 31;

    // ---------- phase 0a: async-stage patch (f32) and f16 qkv weights into LDS ----------
    for (int e = 0; e < 4; ++e) {              // 512 x 16B = 8KB patch (64 tok x 32ch f32)
        int c = tid + 128 * e;
        int tok = c >> 3, ch0 = (c & 7) * 4;
        int pr = reflect_idx(wy * 4 + (tok >> 3));
        int pc = reflect_idx(wx * 4 + (tok & 7));
        int goff = ((((t * Hh + pr) * Ww) + pc) * Cc + ch0) * 4;
        async_copy16(lds_base + OFF_ST + c * 16, x, goff);
    }
    for (int e = 0; e < 3; ++e) {              // 384 x 16B = 6KB qkv weights (f16, padded rows)
        int c = tid + 128 * e;
        int row = c >> 2, k0 = (c & 3) * 8;
        async_copy16(lds_base + OFF_WQ + (row * 40 + k0) * 2, wq16, (row * 32 + k0) * 2);
    }
    for (int e = 0; e < 8; ++e) {              // 32x32 proj weights (small; VALU convert)
        int flat = tid + 128 * e;
        pw[(flat >> 5) * 36 + (flat & 31)] = (_Float16)proj_w[flat];
    }
    if (t == 0 && tid < 64) {                  // coverage counts (t-independent)
        int pr = reflect_idx(wy * 4 + (tid >> 3));
        int pc = reflect_idx(wx * 4 + (tid & 7));
        atomicAdd(&cnt[pr * Ww + pc], 1.0f);
    }
    wait_async0();
    __syncthreads();

    // ---------- phase 0b: convert staged f32 patch -> f16 A-operand layout ----------
    for (int e = 0; e < 16; ++e) {
        int flat = tid + 128 * e;              // 2048
        int tok = flat >> 5, ch = flat & 31;
        ph[tok * 40 + ch] = (_Float16)stg[flat];
    }
    __syncthreads();

    // ---------- phase 1: QKV = P @ Wqkv^T + b   (M=64, N=96, K=32) ----------
    for (int it = 0; it < 6; ++it) {
        int tile = wave + 4 * it;              // 24 tiles over 4 waves (scalar)
        int mt = tile / 6, nt = tile % 6;
        v16h a, bm;
        #pragma unroll
        for (int idx = 0; idx < 16; ++idx) {
            int ka = 16 * (idx >> 3) + 8 * hv + (idx & 7);
            a[idx]  = ph[(16 * mt + lm) * 40 + ka];
            int kb = 16 * (idx >> 3) + 8 * hv + (idx & 7);
            bm[idx] = wq[(16 * nt + lm) * 40 + kb];
        }
        v8f c = {};
        c = __builtin_amdgcn_wmma_f32_16x16x32_f16(false, a, false, bm, (short)0, c, false, false);
        int col = 16 * nt + lm;
        float bias = qkv_b[col];
        if (nt < 2) {                          // q columns (scalar branch)
            #pragma unroll
            for (int j = 0; j < 8; ++j) {
                int row = 16 * mt + j + 8 * hv;
                qk[row * 67 + col] = (_Float16)((c[j] + bias) * 0.25f);   // q * hd^-0.5
            }
        } else if (nt < 4) {                   // k columns
            #pragma unroll
            for (int j = 0; j < 8; ++j) {
                int row = 16 * mt + j + 8 * hv;
                qk[row * 67 + col] = (_Float16)(c[j] + bias);
            }
        } else {                               // v columns, stored transposed [ch][tok]
            #pragma unroll
            for (int j = 0; j < 8; ++j) {
                int row = 16 * mt + j + 8 * hv;
                vT[(col - 64) * 68 + row] = (_Float16)(c[j] + bias);
            }
        }
    }
    __syncthreads();

    // ---------- phase 2: logits = q @ k^T per head (K=16 zero-padded to 32) ----------
    for (int it = 0; it < 8; ++it) {
        int tile = wave + 4 * it;              // 32 tiles: 2 heads x 4x4 (scalar)
        int head = tile >> 4, mt = (tile >> 2) & 3, nt = tile & 3;
        v16h a, bm;
        #pragma unroll
        for (int idx = 0; idx < 16; ++idx) {
            _Float16 av = (_Float16)0.f;
            if (idx < 8)
                av = qk[(16 * mt + lm) * 67 + head * 16 + 8 * hv + idx];
            a[idx] = av;
            _Float16 bv = (_Float16)0.f;
            if (idx < 8)
                bv = qk[(16 * nt + lm) * 67 + 32 + head * 16 + 8 * hv + idx];
            bm[idx] = bv;
        }
        v8f c = {};
        c = __builtin_amdgcn_wmma_f32_16x16x32_f16(false, a, false, bm, (short)0, c, false, false);
        #pragma unroll
        for (int j = 0; j < 8; ++j) {
            int row = 16 * mt + j + 8 * hv;
            int col = 16 * nt + lm;
            af[(head * 64 + row) * 65 + col] = c[j];
        }
    }
    __syncthreads();

    // ---------- phase 3: bias + softmax (one row per thread, f32) ----------
    {
        int head = tid >> 6, row = tid & 63;
        float lv[64];
        float mx = -3.0e38f;
        #pragma unroll
        for (int m = 0; m < 64; ++m) {
            float v = af[(head * 64 + row) * 65 + m] + table[rindex[row * 64 + m] * 2 + head];
            lv[m] = v;
            mx = fmaxf(mx, v);
        }
        float s = 0.f;
        #pragma unroll
        for (int m = 0; m < 64; ++m) { float e = __expf(lv[m] - mx); lv[m] = e; s += e; }
        float inv = 1.0f / s;
        #pragma unroll
        for (int m = 0; m < 64; ++m)
            ah[(head * 64 + row) * 66 + m] = (_Float16)(lv[m] * inv);
    }
    __syncthreads();

    // ---------- phase 4: O = attn @ V per head (M=64, N=16, K=64 -> two k-steps) ----------
    for (int it = 0; it < 2; ++it) {
        int tile = wave + 4 * it;              // 8 tiles: 2 heads x 4 (scalar)
        int head = tile >> 2, mt = tile & 3;
        v8f c = {};
        #pragma unroll
        for (int kb = 0; kb < 64; kb += 32) {
            v16h a, bm;
            #pragma unroll
            for (int idx = 0; idx < 16; ++idx) {
                int ka = kb + 16 * (idx >> 3) + 8 * hv + (idx & 7);
                a[idx]  = ah[(head * 64 + 16 * mt + lm) * 66 + ka];
                int tok = kb + 16 * (idx >> 3) + 8 * hv + (idx & 7);   // table order, matches A
                bm[idx] = vT[(head * 16 + lm) * 68 + tok];
            }
            c = __builtin_amdgcn_wmma_f32_16x16x32_f16(false, a, false, bm, (short)0, c, false, false);
        }
        #pragma unroll
        for (int j = 0; j < 8; ++j) {
            int row = 16 * mt + j + 8 * hv;
            avh[row * 40 + head * 16 + lm] = (_Float16)c[j];
        }
    }
    __syncthreads();

    // ---------- phase 5: proj + reflect scatter-add (M=64, N=32, K=32) ----------
    for (int it = 0; it < 2; ++it) {
        int tile = wave + 4 * it;              // 8 tiles: 4x2 (scalar)
        int mt = tile >> 1, nt = tile & 1;
        v16h a, bm;
        #pragma unroll
        for (int idx = 0; idx < 16; ++idx) {
            int ka = 16 * (idx >> 3) + 8 * hv + (idx & 7);
            a[idx]  = avh[(16 * mt + lm) * 40 + ka];
            int kb = 16 * (idx >> 3) + 8 * hv + (idx & 7);
            bm[idx] = pw[(16 * nt + lm) * 36 + kb];
        }
        v8f c = {};
        c = __builtin_amdgcn_wmma_f32_16x16x32_f16(false, a, false, bm, (short)0, c, false, false);
        #pragma unroll
        for (int j = 0; j < 8; ++j) {
            int l   = 16 * mt + j + 8 * hv;
            int col = 16 * nt + lm;
            float v = c[j] + proj_b[col];
            int pr = reflect_idx(wy * 4 + (l >> 3));
            int pc = reflect_idx(wx * 4 + (l & 7));
            atomicAdd(&out[(((t * Hh) + pr) * Ww + pc) * Cc + col], v);
        }
    }
}

__global__ void prep_weights(const float* __restrict__ qkv_w, _Float16* __restrict__ wq16) {
    int i = blockIdx.x * blockDim.x + threadIdx.x;
    if (i < 96 * 32) wq16[i] = (_Float16)qkv_w[i];
}

__global__ void zero_kernel(float* __restrict__ out, float* __restrict__ cnt, int n) {
    int i = blockIdx.x * blockDim.x + threadIdx.x;
    if (i < n) out[i] = 0.f;
    if (i < Hh * Ww) cnt[i] = 0.f;
}

__global__ void norm_kernel(float* __restrict__ out, const float* __restrict__ cnt, int n) {
    int i = blockIdx.x * blockDim.x + threadIdx.x;
    if (i < n) {
        int pix = (i / Cc) % (Hh * Ww);
        out[i] = out[i] / (cnt[pix] + 1e-10f);
    }
}

extern "C" void kernel_launch(void* const* d_in, const int* in_sizes, int n_in,
                              void* d_out, int out_size, void* d_ws, size_t ws_size,
                              hipStream_t stream) {
    const float* x      = (const float*)d_in[0];
    const float* qkv_w  = (const float*)d_in[1];
    const float* qkv_b  = (const float*)d_in[2];
    const float* table  = (const float*)d_in[3];
    const float* proj_w = (const float*)d_in[4];
    const float* proj_b = (const float*)d_in[5];
    const int*   rindex = (const int*)d_in[6];
    float* out = (float*)d_out;
    float* cnt = (float*)d_ws;                                   // 64KB
    _Float16* wq16 = (_Float16*)((char*)d_ws + WS_WQ16);         // 6144B

    const int n = Tt * Hh * Ww * Cc;
    zero_kernel<<<(n + 255) / 256, 256, 0, stream>>>(out, cnt, n);
    prep_weights<<<12, 256, 0, stream>>>(qkv_w, wq16);
    win_attn_kernel<<<NWIN, 128, SMEM_BYTES, stream>>>(x, qkv_b, table, proj_w, proj_b,
                                                       rindex, wq16, out, cnt);
    norm_kernel<<<(n + 255) / 256, 256, 0, stream>>>(out, cnt, n);
}
